// SparseMoeBlock_65747359367216
// MI455X (gfx1250) — matrix operand order repaired and checked by reading, not compiled, not run on hardware
//
#include <hip/hip_runtime.h>
#include <hip/hip_bf16.h>

#define H    1024
#define FFN  4096
#define NE   8
#define T    2048
#define CHUNK 512      // FFN chunk held in LDS as bf16 activations
#define NCHUNK (FFN / CHUNK)

typedef unsigned int u32;
typedef unsigned short u16;
typedef u32  u32x4  __attribute__((ext_vector_type(4)));
typedef u32  u32x8  __attribute__((ext_vector_type(8)));
typedef __bf16 bf16x16 __attribute__((ext_vector_type(16)));
typedef float f32x8 __attribute__((ext_vector_type(8)));

// Pack two f32 into a bf16 pair (truncation) with a single v_perm_b32:
// D[15:0] = a[31:16], D[31:16] = b[31:16].
// v_perm selector: bytes 0-3 pick from S1, 4-7 from S0 (S0 = first arg).
__device__ __forceinline__ u32 pkt(u32 a_bits, u32 b_bits) {
    return __builtin_amdgcn_perm(b_bits, a_bits, 0x07060302u);
}

// ---- f32 -> bf16 round-to-nearest-even (cold paths only) ----
__device__ __forceinline__ u16 f2bf(float f) {
    u32 u = __float_as_uint(f);
    u32 r = (u + 0x7FFFu + ((u >> 16) & 1u)) >> 16;
    return (u16)r;
}

// A-fragment (16x32 bf16, lane = row m in [0,16), hb selects K sub-bank).
// lanes 0-15 hold K = k..k+15; lanes 16-31 hold
// K = k+16..k+31  -> two contiguous 16B LDS loads.
__device__ __forceinline__ bf16x16 frag_a_lds(const u16* p) {
    u32x4 lo = *(const u32x4*)p;
    u32x4 hi = *(const u32x4*)(p + 8);
    u32x8 v;
    v[0] = lo[0]; v[1] = lo[1]; v[2] = lo[2]; v[3] = lo[3];
    v[4] = hi[0]; v[5] = hi[1]; v[6] = hi[2]; v[7] = hi[3];
    return __builtin_bit_cast(bf16x16, v);
}

// B-fragment (32x16 bf16): lane owns column n = lane&15; holds 16 contiguous
// K values. Loaded from global f32 rows (4x b128 per lane) and packed to
// bf16 with 8 v_perm_b32 (vs ~64 VALU for scalar RNE).
__device__ __forceinline__ bf16x16 frag_b_gl(const float* p) {
    u32x4 q0 = *(const u32x4*)(p);
    u32x4 q1 = *(const u32x4*)(p + 4);
    u32x4 q2 = *(const u32x4*)(p + 8);
    u32x4 q3 = *(const u32x4*)(p + 12);
    u32x8 v;
    v[0] = pkt(q0[0], q0[1]); v[1] = pkt(q0[2], q0[3]);
    v[2] = pkt(q1[0], q1[1]); v[3] = pkt(q1[2], q1[3]);
    v[4] = pkt(q2[0], q2[1]); v[5] = pkt(q2[2], q2[3]);
    v[6] = pkt(q3[0], q3[1]); v[7] = pkt(q3[2], q3[3]);
    return __builtin_bit_cast(bf16x16, v);
}

#define WMMA_BF16(A, B, C) \
    __builtin_amdgcn_wmma_f32_16x16x32_bf16(false, (A), false, (B), (short)0, (C), false, false)

// ---------------- kernel 0: zero y-region of output + expert counters -------
__global__ void moe_zero(float* __restrict__ outY, int n, int* __restrict__ counts) {
    int stride = gridDim.x * blockDim.x;
    for (int i = blockIdx.x * blockDim.x + threadIdx.x; i < n; i += stride)
        outY[i] = 0.f;
    if (blockIdx.x == 0 && threadIdx.x < NE) counts[threadIdx.x] = 0;
}

// ---------------- kernel 1: router (logits, softmax, top-2, scatter) --------
__global__ void moe_router(const float* __restrict__ x, const float* __restrict__ gw,
                           float* __restrict__ logitsOut, int* __restrict__ counts,
                           int* __restrict__ tokList, float* __restrict__ wtList) {
    const int lane = threadIdx.x & 31;
    const int wid  = (blockIdx.x * blockDim.x + threadIdx.x) >> 5;
    const int nw   = (gridDim.x * blockDim.x) >> 5;
    for (int t = wid; t < T; t += nw) {
        float a[NE];
#pragma unroll
        for (int e = 0; e < NE; ++e) a[e] = 0.f;
        const float* xr = x + (size_t)t * H;
        for (int i = lane; i < H; i += 32) {
            float xv = xr[i];
#pragma unroll
            for (int e = 0; e < NE; ++e) a[e] += xv * gw[e * H + i];
        }
#pragma unroll
        for (int e = 0; e < NE; ++e)
            for (int off = 16; off > 0; off >>= 1)
                a[e] += __shfl_xor(a[e], off, 32);
        if (lane == 0) {
            float mx = a[0];
#pragma unroll
            for (int e = 1; e < NE; ++e) mx = fmaxf(mx, a[e]);
            float p[NE];
#pragma unroll
            for (int e = 0; e < NE; ++e) p[e] = __expf(a[e] - mx);
            int a0 = 0;
#pragma unroll
            for (int e = 1; e < NE; ++e) if (p[e] > p[a0]) a0 = e;
            int a1 = (a0 == 0) ? 1 : 0;
#pragma unroll
            for (int e = 0; e < NE; ++e) if (e != a0 && p[e] > p[a1]) a1 = e;
            float den = p[a0] + p[a1];
#pragma unroll
            for (int e = 0; e < NE; ++e) logitsOut[t * NE + e] = a[e];
            int s0 = atomicAdd(&counts[a0], 1);
            tokList[a0 * T + s0] = t; wtList[a0 * T + s0] = p[a0] / den;
            int s1 = atomicAdd(&counts[a1], 1);
            tokList[a1 * T + s1] = t; wtList[a1 * T + s1] = p[a1] / den;
        }
    }
}

// ---------------- kernel 2: per-expert fused w13 -> silu*u -> w2 GEMM -------
// Block = 256 threads (8 waves), one 16-token tile of one expert.
__global__ __launch_bounds__(256)
void moe_gemm(const float* __restrict__ x, const float* __restrict__ w13,
              const float* __restrict__ w2, const int* __restrict__ counts,
              const int* __restrict__ tokList, const float* __restrict__ wtList,
              float* __restrict__ outY) {
    __shared__ __align__(16) u16 sX[16 * H];        // x tile, bf16   (32 KB)
    __shared__ __align__(16) u16 sAct[16 * CHUNK];  // act chunk bf16 (16 KB)
    __shared__ float sWt[16];
    __shared__ int   sTok[16];

    const int e    = blockIdx.y;
    const int tile = blockIdx.x;
    const int cnt  = counts[e];
    if (tile * 16 >= cnt) return;

    if (threadIdx.x < 16) {
        int idx = tile * 16 + threadIdx.x;
        if (idx < cnt) { sTok[threadIdx.x] = tokList[e * T + idx];
                         sWt[threadIdx.x] = wtList[e * T + idx]; }
        else           { sTok[threadIdx.x] = 0; sWt[threadIdx.x] = 0.f; }
    }
    __syncthreads();

    // Gather 16 token rows, convert f32 -> bf16 once (v_perm pack).
    for (int idx = threadIdx.x; idx < 16 * (H / 2); idx += 256) {
        int i2  = idx << 1;
        int row = i2 >> 10;            // /H
        int col = i2 & (H - 1);
        const u32* px = (const u32*)(x + (size_t)sTok[row] * H + col);
        ((u32*)sX)[idx] = pkt(px[0], px[1]);
    }
    __syncthreads();

    const int lane = threadIdx.x & 31;
    const int wv   = threadIdx.x >> 5;
    const int nl   = lane & 15;
    const int hb   = lane >> 4;

    const float* w13e = w13 + (size_t)e * (2 * FFN) * H;
    const float* w2e  = w2  + (size_t)e * H * FFN;

    f32x8 acc[8];
#pragma unroll
    for (int i = 0; i < 8; ++i) acc[i] = f32x8{0.f,0.f,0.f,0.f,0.f,0.f,0.f,0.f};

    for (int chunk = 0; chunk < NCHUNK; ++chunk) {
        const int fbase = chunk * CHUNK;

        // ---- Phase A: act[16, CHUNK] = silu(x@w1^T) * (x@w3^T) * route_wt
        // Process column-tiles in pairs so each A fragment feeds 4 WMMAs.
#pragma unroll
        for (int np = 0; np < 2; ++np) {
            const int lc0 = wv * 64 + (2 * np) * 16;       // local col bases
            const int lc1 = lc0 + 16;
            const float* g0 = w13e + (size_t)(fbase + lc0 + nl) * H;
            const float* u0 = w13e + (size_t)(FFN + fbase + lc0 + nl) * H;
            f32x8 cg0 = {0.f,0.f,0.f,0.f,0.f,0.f,0.f,0.f};
            f32x8 cu0 = {0.f,0.f,0.f,0.f,0.f,0.f,0.f,0.f};
            f32x8 cg1 = {0.f,0.f,0.f,0.f,0.f,0.f,0.f,0.f};
            f32x8 cu1 = {0.f,0.f,0.f,0.f,0.f,0.f,0.f,0.f};
#pragma unroll 2
            for (int kb = 0; kb < H / 32; ++kb) {
                const int k = kb * 32;
                bf16x16 a = frag_a_lds(sX + nl * H + k + hb * 16);
                cg0 = WMMA_BF16(a, frag_b_gl(g0 + k + hb * 16), cg0);
                cu0 = WMMA_BF16(a, frag_b_gl(u0 + k + hb * 16), cu0);
                cg1 = WMMA_BF16(a, frag_b_gl(g0 + (size_t)16 * H + k + hb * 16), cg1);
                cu1 = WMMA_BF16(a, frag_b_gl(u0 + (size_t)16 * H + k + hb * 16), cu1);
            }
#pragma unroll
            for (int r = 0; r < 8; ++r) {
                const int m = r + 8 * hb;
                float wt = sWt[m];
                float ga = cg0[r], ua = cu0[r];
                float gb = cg1[r], ub = cu1[r];
                sAct[m * CHUNK + lc0 + nl] = f2bf((ga / (1.f + __expf(-ga))) * ua * wt);
                sAct[m * CHUNK + lc1 + nl] = f2bf((gb / (1.f + __expf(-gb))) * ub * wt);
            }
        }
        __syncthreads();

        // ---- Phase B: y[16, H] += act @ w2^T.
        // K outer, H-tile inner: one A fragment feeds 8 WMMAs.
        const float* brow = w2e + (size_t)(wv * 128 + nl) * FFN + fbase;
#pragma unroll 2
        for (int kb = 0; kb < CHUNK / 32; ++kb) {
            const int k = kb * 32;
            bf16x16 a = frag_a_lds(sAct + nl * CHUNK + k + hb * 16);
#pragma unroll
            for (int ht = 0; ht < 8; ++ht) {
                acc[ht] = WMMA_BF16(a, frag_b_gl(brow + (size_t)ht * 16 * FFN + k + hb * 16),
                                    acc[ht]);
            }
        }
        __syncthreads();
    }

    // Each token belongs to exactly 2 experts -> accumulate with f32 atomics.
#pragma unroll
    for (int ht = 0; ht < 8; ++ht) {
        const int h = wv * 128 + ht * 16 + nl;
#pragma unroll
        for (int r = 0; r < 8; ++r) {
            const int m = r + 8 * hb;
            atomicAdd(&outY[(size_t)sTok[m] * H + h], acc[ht][r]);
        }
    }
}

extern "C" void kernel_launch(void* const* d_in, const int* in_sizes, int n_in,
                              void* d_out, int out_size, void* d_ws, size_t ws_size,
                              hipStream_t stream) {
    const float* x   = (const float*)d_in[0];   // [T, H]
    const float* gw  = (const float*)d_in[1];   // [E, H]
    const float* w13 = (const float*)d_in[2];   // [E, 2F, H]
    const float* w2  = (const float*)d_in[3];   // [E, H, F]

    float* outY      = (float*)d_out;                    // [T, H]
    float* outLogits = outY + (size_t)T * H;             // [T, E]

    // workspace: counts[16 pad] | tokList[E*T] | wtList[E*T]  (~131 KB)
    int*   counts  = (int*)d_ws;
    int*   tokList = counts + 16;
    float* wtList  = (float*)(tokList + NE * T);

    moe_zero<<<512, 256, 0, stream>>>(outY, T * H, counts);
    moe_router<<<64, 256, 0, stream>>>(x, gw, outLogits, counts, tokList, wtList);
    dim3 grid(T / 16, NE);
    moe_gemm<<<grid, 256, 0, stream>>>(x, w13, w2, counts, tokList, wtList, outY);
}
